// Decoder_70858370449846
// MI455X (gfx1250) — matrix operand order repaired and checked by reading, not compiled, not run on hardware
//
#include <hip/hip_runtime.h>
#include <hip/hip_bf16.h>

// ---------------------------------------------------------------------------
// LAS decoder for MI455X (gfx1250, wave32).
//  - LSTM gate GEMMs in bf16 via v_wmma_f32_16x16x32_bf16 (f32 accumulate)
//  - weights packed once per call into WMMA B-fragment tile layout
//  - attention tail (energy/softmax/ctx/logits/embed-copy) fused into one
//    LDS-resident kernel per step to shorten the 300-step latency chain
// ---------------------------------------------------------------------------

typedef __attribute__((ext_vector_type(16))) __bf16 v16bf;
typedef __attribute__((ext_vector_type(8)))  __bf16 v8bf;
typedef __attribute__((ext_vector_type(8)))  float  v8f;

#define T_  2048
#define N_  64
#define L_  300
#define V_  35
#define E_  256
#define H_  512
#define KS_ 128
#define VS_ 128

__device__ __forceinline__ float sigm(float x) { return 1.0f / (1.0f + __expf(-x)); }

// ---------------------------------------------------------------------------
// Pack W = [Wih | Whh] (row-major, Ncols x (Kih+Khh) f32) into bf16 WMMA
// B-fragment tiles.  Tile = 16 cols x 32 K, lane-major: lane (0..31) holds
// 16 contiguous bf16:  col = lane&15,  k = ktile*32 + (lane>>4)*16 + j.
// ---------------------------------------------------------------------------
__global__ void packw_kernel(const float* __restrict__ Wih, int Kih,
                             const float* __restrict__ Whh, int Khh,
                             int Ncols, __bf16* __restrict__ dst) {
    const int Ka = Kih + Khh;
    const size_t total = (size_t)Ncols * (size_t)Ka;
    size_t idx = (size_t)blockIdx.x * blockDim.x + threadIdx.x;
    if (idx >= total) return;
    const int j    = (int)(idx & 15);
    const int lane = (int)((idx >> 4) & 31);
    const size_t tile = idx >> 9;              // nt*KT + kt
    const int KT = Ka >> 5;
    const int nt = (int)(tile / KT);
    const int kt = (int)(tile % KT);
    const int ncol = nt * 16 + (lane & 15);
    const int k    = kt * 32 + ((lane >> 4) << 4) + j;
    float v = (k < Kih) ? Wih[(size_t)ncol * Kih + k]
                        : Whh[(size_t)ncol * Khh + (k - Kih)];
    dst[idx] = (__bf16)v;
}

// ---------------------------------------------------------------------------
// GEMM: C(64 x Ncols, f32) = A(64 x Ka, bf16 row-major) @ Bpk^T
// block = 128 threads = 4 waves; wave w owns M-tile w; grid.x = Ncols/16.
// ---------------------------------------------------------------------------
__global__ __launch_bounds__(128)
void gemm_bf16_wmma_kernel(const __bf16* __restrict__ A,
                           const __bf16* __restrict__ Bpk,
                           float* __restrict__ C,
                           int Ka, int Ncols) {
    const int nt    = blockIdx.x;
    const int wave  = threadIdx.x >> 5;
    const int lane  = threadIdx.x & 31;
    const int row   = (wave << 4) + (lane & 15);
    const int khalf = lane >> 4;
    const int KT    = Ka >> 5;

    v8f acc = {};
    const __bf16* arow = A + (size_t)row * Ka + khalf * 16;
    const __bf16* bt   = Bpk + (size_t)nt * KT * 512 + lane * 16;

    for (int kt = 0; kt < KT; ++kt) {
        __builtin_prefetch((const void*)(bt + 4 * 512), 0, 3);  // near-cache prefetch
        v8bf a0 = *(const v8bf*)(arow);
        v8bf a1 = *(const v8bf*)(arow + 8);
        v16bf av;
#pragma unroll
        for (int i = 0; i < 8; ++i) { av[i] = a0[i]; av[8 + i] = a1[i]; }
        v16bf bv = *(const v16bf*)(bt);
        acc = __builtin_amdgcn_wmma_f32_16x16x32_bf16(
            false, av, false, bv, (short)0, acc, false, false);
        arow += 32;
        bt   += 512;
    }
    const int ncol = nt * 16 + (lane & 15);
#pragma unroll
    for (int r = 0; r < 8; ++r) {
        const int m = (wave << 4) + (khalf << 3) + r;
        C[(size_t)m * Ncols + ncol] = acc[r];
    }
}

// ---------------------------------------------------------------------------
// Fused LSTM elementwise: gates (64 x 4H, f32) + biases -> c,h update.
// ---------------------------------------------------------------------------
__global__ void lstm_ew_kernel(const float* __restrict__ gates,
                               const float* __restrict__ bih,
                               const float* __restrict__ bhh,
                               float* __restrict__ cstate, int Hdim,
                               __bf16* __restrict__ hdst1, int stride1,
                               __bf16* __restrict__ hdst2, int stride2,
                               float* __restrict__ hf32) {
    int idx = blockIdx.x * blockDim.x + threadIdx.x;
    if (idx >= N_ * Hdim) return;
    const int n = idx / Hdim;
    const int j = idx % Hdim;
    const float* gr = gates + (size_t)n * 4 * Hdim;
    float gi = sigm (gr[j]            + bih[j]            + bhh[j]);
    float gf = sigm (gr[Hdim + j]     + bih[Hdim + j]     + bhh[Hdim + j]);
    float gg = tanhf(gr[2 * Hdim + j] + bih[2 * Hdim + j] + bhh[2 * Hdim + j]);
    float go = sigm (gr[3 * Hdim + j] + bih[3 * Hdim + j] + bhh[3 * Hdim + j]);
    float c = gf * cstate[idx] + gi * gg;
    cstate[idx] = c;
    float h = go * tanhf(c);
    __bf16 hb = (__bf16)h;
    hdst1[(size_t)n * stride1 + j] = hb;
    if (hdst2) hdst2[(size_t)n * stride2 + j] = hb;
    if (hf32)  hf32[idx] = h;
}

// embeds_bf[l][n][e] = bf16(emb_W[text[n][l]][e])
__global__ void embed_setup_kernel(const int* __restrict__ text,
                                   const float* __restrict__ embW,
                                   __bf16* __restrict__ embB) {
    size_t idx = (size_t)blockIdx.x * blockDim.x + threadIdx.x;
    if (idx >= (size_t)L_ * N_ * E_) return;
    const int e = (int)(idx & (E_ - 1));
    const int n = (int)((idx >> 8) & (N_ - 1));
    const int l = (int)(idx >> 14);
    const int tok = text[(size_t)n * L_ + l];
    embB[idx] = (__bf16)embW[(size_t)tok * E_ + e];
}

// copy embeds[0] into A1[:, 0:256] (steps >=1 handled by attention epilogue)
__global__ void a1_embed_kernel(const __bf16* __restrict__ embB,
                                __bf16* __restrict__ A1, int l) {
    int idx = blockIdx.x * blockDim.x + threadIdx.x;   // 64*256
    const int n = idx >> 8;
    const int e = idx & (E_ - 1);
    A1[(size_t)n * 896 + e] = embB[((size_t)l * N_ + n) * E_ + e];
}

// ---------------------------------------------------------------------------
// Fused attention tail: one block per batch row n.
//   energy[t] = key[t][n][:].h3[n][:]  (wave-per-t, shuffle reduce, LDS store)
//   masked softmax over T in LDS; write att output
//   ctx[e] = sum_t p[t]*vals[t][n][e]  (split across 2 half-ranges of t)
//   logits[v] = [h3|ctx].emb_W[v] + b_out[v] -> predictions
//   epilogue: copy embeds[l+1][n] into A1[:,0:256] for the next step
// ---------------------------------------------------------------------------
__global__ __launch_bounds__(256)
void attention_fused_kernel(const float* __restrict__ key,
                            const float* __restrict__ vals,
                            const float* __restrict__ h3f,
                            const int*   __restrict__ x_lens,
                            const __bf16* __restrict__ embB,
                            const float* __restrict__ embW,
                            const float* __restrict__ b_out,
                            __bf16* __restrict__ A1,
                            float* __restrict__ att_out,
                            float* __restrict__ pred,
                            int l) {
    __shared__ float sE[T_];        // energies -> probabilities
    __shared__ float red[256];
    __shared__ float csh[256];
    __shared__ float sh3[KS_];
    __shared__ float sctx[VS_];

    const int n    = blockIdx.x;
    const int tid  = threadIdx.x;
    const int wave = tid >> 5;
    const int lane = tid & 31;
    const int len  = x_lens[n];

    if (tid < KS_) sh3[tid] = h3f[(size_t)n * KS_ + tid];

    // ---- energy: one wave per t (strided by 8 waves) ----
    const float4 h4 = *(const float4*)(h3f + (size_t)n * KS_ + lane * 4);
    for (int t = wave; t < T_; t += 8) {
        const float4 k4 = *(const float4*)(key + ((size_t)t * N_ + n) * KS_ + lane * 4);
        float s = k4.x * h4.x + k4.y * h4.y + k4.z * h4.z + k4.w * h4.w;
#pragma unroll
        for (int off = 16; off; off >>= 1) s += __shfl_xor(s, off, 32);
        if (lane == 0) sE[t] = (t >= len) ? -1e9f : s;
    }
    __syncthreads();

    // ---- softmax over T_ (each thread owns 8 consecutive t) ----
    float m = -3.4e38f;
#pragma unroll
    for (int i = 0; i < 8; ++i) m = fmaxf(m, sE[tid * 8 + i]);
    red[tid] = m; __syncthreads();
    for (int s = 128; s; s >>= 1) {
        if (tid < s) red[tid] = fmaxf(red[tid], red[tid + s]);
        __syncthreads();
    }
    m = red[0]; __syncthreads();

    float sum = 0.0f;
#pragma unroll
    for (int i = 0; i < 8; ++i) {
        float p = __expf(sE[tid * 8 + i] - m);
        sE[tid * 8 + i] = p;
        sum += p;
    }
    red[tid] = sum; __syncthreads();
    for (int s = 128; s; s >>= 1) {
        if (tid < s) red[tid] += red[tid + s];
        __syncthreads();
    }
    const float inv = 1.0f / red[0];
    float* ao = att_out + ((size_t)n * L_ + l) * T_;
#pragma unroll
    for (int i = 0; i < 8; ++i) {
        float p = sE[tid * 8 + i] * inv;
        sE[tid * 8 + i] = p;
        ao[tid * 8 + i] = p;
    }
    __syncthreads();

    // ---- context: thread (e, half) accumulates half the t range ----
    {
        const int e    = tid & (VS_ - 1);
        const int half = tid >> 7;
        const float* vb = vals + (size_t)n * VS_ + e;   // + t*N_*VS_
        float acc = 0.0f;
        const int t0 = half * (T_ / 2);
#pragma unroll 4
        for (int t = t0; t < t0 + T_ / 2; ++t)
            acc += sE[t] * vb[(size_t)t * (N_ * VS_)];
        csh[tid] = acc;
    }
    __syncthreads();
    if (tid < VS_) {
        float c = csh[tid] + csh[tid + VS_];
        sctx[tid] = c;
        A1[(size_t)n * 896 + 256 + tid] = (__bf16)c;    // ctx -> next LSTM1 input
    }
    __syncthreads();

    // ---- logits -> predictions ----
    if (tid < V_) {
        float acc = b_out[tid];
        const float* w = embW + (size_t)tid * E_;
#pragma unroll 4
        for (int k = 0; k < KS_; ++k) acc += sh3[k] * w[k];
#pragma unroll 4
        for (int k = 0; k < VS_; ++k) acc += sctx[k] * w[KS_ + k];
        pred[((size_t)n * L_ + l) * V_ + tid] = acc;
    }

    // ---- epilogue: next step's embedding into A1[:,0:256] ----
    if (l + 1 < L_)
        A1[(size_t)n * 896 + tid] = embB[((size_t)(l + 1) * N_ + n) * E_ + tid];
}

__global__ void zero_ws_kernel(unsigned int* __restrict__ p, int nWords) {
    int i = blockIdx.x * blockDim.x + threadIdx.x;
    if (i < nWords) p[i] = 0u;
}

// ---------------------------------------------------------------------------
extern "C" void kernel_launch(void* const* d_in, const int* in_sizes, int n_in,
                              void* d_out, int out_size, void* d_ws, size_t ws_size,
                              hipStream_t stream) {
    const float* key    = (const float*)d_in[0];
    const float* values = (const float*)d_in[1];
    const int*   x_lens = (const int*)  d_in[2];
    const int*   text   = (const int*)  d_in[3];
    const float* emb_W  = (const float*)d_in[4];
    const float* W_ih1  = (const float*)d_in[5];
    const float* W_hh1  = (const float*)d_in[6];
    const float* b_ih1  = (const float*)d_in[7];
    const float* b_hh1  = (const float*)d_in[8];
    const float* W_ih2  = (const float*)d_in[9];
    const float* W_hh2  = (const float*)d_in[10];
    const float* b_ih2  = (const float*)d_in[11];
    const float* b_hh2  = (const float*)d_in[12];
    const float* W_ih3  = (const float*)d_in[13];
    const float* W_hh3  = (const float*)d_in[14];
    const float* b_ih3  = (const float*)d_in[15];
    const float* b_hh3  = (const float*)d_in[16];
    const float* b_out  = (const float*)d_in[17];

    float* pred = (float*)d_out;                               // (N,L,V)
    float* att  = (float*)d_out + (size_t)N_ * L_ * V_;        // (N,L,T)

    // ---- workspace carve-up (256B aligned) ----
    char* ws = (char*)d_ws;
    size_t off = 0;
    auto carve = [&](size_t bytes) -> char* {
        off = (off + 255) & ~(size_t)255;
        char* p = ws + off;
        off += bytes;
        return p;
    };
    __bf16* Wc1  = (__bf16*)carve((size_t)2048 * 896  * 2);
    __bf16* Wc2  = (__bf16*)carve((size_t)2048 * 1024 * 2);
    __bf16* Wc3  = (__bf16*)carve((size_t)512  * 640  * 2);
    __bf16* embB = (__bf16*)carve((size_t)L_ * N_ * E_ * 2);
    __bf16* A1   = (__bf16*)carve((size_t)N_ * 896  * 2);
    __bf16* A2   = (__bf16*)carve((size_t)N_ * 1024 * 2);
    __bf16* A3   = (__bf16*)carve((size_t)N_ * 640  * 2);
    float*  gates = (float*)carve((size_t)N_ * 2048 * 4);
    float*  c1    = (float*)carve((size_t)N_ * H_  * 4);
    float*  c2    = (float*)carve((size_t)N_ * H_  * 4);
    float*  c3    = (float*)carve((size_t)N_ * KS_ * 4);
    float*  h3f   = (float*)carve((size_t)N_ * KS_ * 4);
    (void)ws_size; (void)in_sizes; (void)n_in; (void)out_size;

    // ---- one-time-per-call setup ----
    auto zero = [&](void* p, size_t bytes) {
        int w = (int)(bytes / 4);
        zero_ws_kernel<<<(w + 255) / 256, 256, 0, stream>>>((unsigned int*)p, w);
    };
    zero(A1, (size_t)N_ * 896 * 2);   zero(A2, (size_t)N_ * 1024 * 2);
    zero(A3, (size_t)N_ * 640 * 2);
    zero(c1, (size_t)N_ * H_ * 4);    zero(c2, (size_t)N_ * H_ * 4);
    zero(c3, (size_t)N_ * KS_ * 4);   zero(h3f, (size_t)N_ * KS_ * 4);

    packw_kernel<<<((size_t)2048 * 896  + 255) / 256, 256, 0, stream>>>(W_ih1, 384, W_hh1, 512, 2048, Wc1);
    packw_kernel<<<((size_t)2048 * 1024 + 255) / 256, 256, 0, stream>>>(W_ih2, 512, W_hh2, 512, 2048, Wc2);
    packw_kernel<<<((size_t)512  * 640  + 255) / 256, 256, 0, stream>>>(W_ih3, 512, W_hh3, 128, 512,  Wc3);
    embed_setup_kernel<<<((size_t)L_ * N_ * E_ + 255) / 256, 256, 0, stream>>>(text, emb_W, embB);
    a1_embed_kernel<<<(N_ * E_) / 256, 256, 0, stream>>>(embB, A1, 0);

    // ---- 300 sequential decode steps ----
    for (int l = 0; l < L_; ++l) {
        // LSTM1: gates = [emb|ctx|h1] @ Wc1^T
        gemm_bf16_wmma_kernel<<<2048 / 16, 128, 0, stream>>>(A1, Wc1, gates, 896, 2048);
        lstm_ew_kernel<<<(N_ * H_ + 255) / 256, 256, 0, stream>>>(
            gates, b_ih1, b_hh1, c1, H_, A1 + 384, 896, A2, 1024, nullptr);

        // LSTM2: gates = [h1|h2] @ Wc2^T
        gemm_bf16_wmma_kernel<<<2048 / 16, 128, 0, stream>>>(A2, Wc2, gates, 1024, 2048);
        lstm_ew_kernel<<<(N_ * H_ + 255) / 256, 256, 0, stream>>>(
            gates, b_ih2, b_hh2, c2, H_, A2 + 512, 1024, A3, 640, nullptr);

        // LSTM3: gates = [h2|h3] @ Wc3^T
        gemm_bf16_wmma_kernel<<<512 / 16, 128, 0, stream>>>(A3, Wc3, gates, 640, 512);
        lstm_ew_kernel<<<(N_ * KS_ + 255) / 256, 256, 0, stream>>>(
            gates, b_ih3, b_hh3, c3, KS_, A3 + 512, 640, nullptr, 0, h3f);

        // fused attention + softmax + context + logits + next-embed
        attention_fused_kernel<<<N_, 256, 0, stream>>>(
            key, values, h3f, x_lens, embB, emb_W, b_out, A1, att, pred, l);
    }
}
